// outconv_47021301957231
// MI455X (gfx1250) — matrix operand order repaired and checked by reading, not compiled, not run on hardware
//
#include <hip/hip_runtime.h>
#include <hip/hip_bf16.h>

#define IMG_H 512
#define IMG_W 512
#define HW    (IMG_H * IMG_W)
#define CIN   64
#define KDIM  576              // 9 taps * 64 cin
#define NTAP  9

typedef _Float16 v16h __attribute__((ext_vector_type(16)));
typedef _Float16 v8h  __attribute__((ext_vector_type(8)));
typedef float    v8f  __attribute__((ext_vector_type(8)));

// ---------------------------------------------------------------------------
// Prep: transpose dcn weights to f16 Wh[o][kappa], kappa = tap*64 + cin,
// and fold conv-bias + batchnorm into per-channel scale/shift.
// ---------------------------------------------------------------------------
__global__ __launch_bounds__(256) void prep_kernel(
    const float* __restrict__ dcn1_w, const float* __restrict__ dcn1_b,
    const float* __restrict__ g1, const float* __restrict__ b1,
    const float* __restrict__ m1, const float* __restrict__ v1,
    const float* __restrict__ dcn2_w, const float* __restrict__ dcn2_b,
    const float* __restrict__ g2, const float* __restrict__ b2,
    const float* __restrict__ m2, const float* __restrict__ v2,
    _Float16* __restrict__ Wh1, _Float16* __restrict__ Wh2,
    float* __restrict__ sc1, float* __restrict__ sh1,
    float* __restrict__ sc2, float* __restrict__ sh2)
{
    int tid = blockIdx.x * blockDim.x + threadIdx.x;
    if (tid < 64 * KDIM) {
        int o   = tid / KDIM;
        int kk  = tid % KDIM;
        int tap = kk / 64;
        int cin = kk % 64;
        Wh1[tid] = (_Float16)dcn1_w[(o * 64 + cin) * 9 + tap];
        Wh2[tid] = (_Float16)dcn2_w[(o * 64 + cin) * 9 + tap];
    }
    if (tid < 64) {
        float inv1 = g1[tid] * rsqrtf(v1[tid] + 1e-5f);
        sc1[tid] = inv1;
        sh1[tid] = b1[tid] + (dcn1_b[tid] - m1[tid]) * inv1;
        float inv2 = g2[tid] * rsqrtf(v2[tid] + 1e-5f);
        sc2[tid] = inv2;
        sh2[tid] = b2[tid] + (dcn2_b[tid] - m2[tid]) * inv2;
    }
}

// ---------------------------------------------------------------------------
// Offset conv: 3x3, 64 -> 18 channels, pad 1. One thread per pixel; weight
// addresses are thread-uniform -> scalar loads.
// ---------------------------------------------------------------------------
__global__ __launch_bounds__(256) void offconv_kernel(
    const float* __restrict__ in, const float* __restrict__ w,
    const float* __restrict__ bias, float* __restrict__ out)
{
    int pix = blockIdx.x * blockDim.x + threadIdx.x;
    int y = pix >> 9;
    int x = pix & 511;

    float acc[18];
#pragma unroll
    for (int co = 0; co < 18; ++co) acc[co] = bias[co];

    for (int cin = 0; cin < CIN; ++cin) {
        const float* ip = in + (size_t)cin * HW;
        float v[9];
#pragma unroll
        for (int t = 0; t < 9; ++t) {
            int yy = y + t / 3 - 1;
            int xx = x + t % 3 - 1;
            bool ok = (yy >= 0) & (yy < IMG_H) & (xx >= 0) & (xx < IMG_W);
            int yc = min(max(yy, 0), IMG_H - 1);
            int xc = min(max(xx, 0), IMG_W - 1);
            float val = ip[yc * IMG_W + xc];
            v[t] = ok ? val : 0.0f;
        }
        const float* wp = w + cin * 9;  // w[(co*64+cin)*9 + t] = w[co*576 + cin*9 + t]
#pragma unroll
        for (int co = 0; co < 18; ++co) {
#pragma unroll
            for (int t = 0; t < 9; ++t)
                acc[co] = fmaf(v[t], wp[co * 576 + t], acc[co]);
        }
    }
#pragma unroll
    for (int co = 0; co < 18; ++co)
        out[(size_t)co * HW + pix] = acc[co];
}

// ---------------------------------------------------------------------------
// Deformable conv + BN + ReLU via WMMA.
// One wave handles 16 consecutive pixels of one row and all 64 out channels:
//     D[64,16] = Wh[64,576] x S[576,16], S = bilinear samples (f16).
// K order: kappa = tap*64 + cin; each 32-K block shares one tap.
// ---------------------------------------------------------------------------
__global__ __launch_bounds__(256) void deform_wmma_kernel(
    const float* __restrict__ in, const float* __restrict__ off,
    const _Float16* __restrict__ Wh, const float* __restrict__ sc,
    const float* __restrict__ sh, float* __restrict__ out)
{
    const int lane  = threadIdx.x & 31;
    const int wid   = (blockIdx.x * blockDim.x + threadIdx.x) >> 5;
    const int p0    = wid * 16;               // tile start (row aligned: 512 % 16 == 0)
    const int py    = p0 >> 9;
    const int px    = (p0 & 511) + (lane & 15);
    const int khalf = lane >> 4;              // 0 = low 16 K of block, 1 = high 16
    const int pix   = py * IMG_W + px;

    // warm the 72KB weight table toward WGP$/L2 (global_prefetch_b8)
    __builtin_prefetch((const char*)Wh + (size_t)lane * 2304, 0, 0);

    v8f acc[4] = {};

    for (int tap = 0; tap < NTAP; ++tap) {
        const int ky = tap / 3 - 1;
        const int kx = tap % 3 - 1;
        float dy = off[(size_t)(2 * tap) * HW + pix];
        float dx = off[(size_t)(2 * tap + 1) * HW + pix];
        float sy = (float)(py + ky) + dy;
        float sx = (float)(px + kx) + dx;
        float fy = floorf(sy), fx = floorf(sx);
        int iy0 = (int)fy, ix0 = (int)fx;
        float wy = sy - fy, wx = sx - fx;

        bool y0ok = (iy0 >= 0) & (iy0 < IMG_H);
        bool y1ok = (iy0 + 1 >= 0) & (iy0 + 1 < IMG_H);
        bool x0ok = (ix0 >= 0) & (ix0 < IMG_W);
        bool x1ok = (ix0 + 1 >= 0) & (ix0 + 1 < IMG_W);
        float w00 = (1.0f - wy) * (1.0f - wx) * ((y0ok & x0ok) ? 1.0f : 0.0f);
        float w01 = (1.0f - wy) * wx          * ((y0ok & x1ok) ? 1.0f : 0.0f);
        float w10 = wy * (1.0f - wx)          * ((y1ok & x0ok) ? 1.0f : 0.0f);
        float w11 = wy * wx                   * ((y1ok & x1ok) ? 1.0f : 0.0f);

        int yc0 = min(max(iy0, 0), IMG_H - 1);
        int yc1 = min(max(iy0 + 1, 0), IMG_H - 1);
        int xc0 = min(max(ix0, 0), IMG_W - 1);
        int xc1 = min(max(ix0 + 1, 0), IMG_W - 1);
        int i00 = yc0 * IMG_W + xc0;
        int i01 = yc0 * IMG_W + xc1;
        int i10 = yc1 * IMG_W + xc0;
        int i11 = yc1 * IMG_W + xc1;

#pragma unroll
        for (int half = 0; half < 2; ++half) {
            const int kb   = tap * 2 + half;
            const int cin0 = half * 32 + khalf * 16;
            const float* q00 = in + (size_t)cin0 * HW + i00;
            const float* q01 = in + (size_t)cin0 * HW + i01;
            const float* q10 = in + (size_t)cin0 * HW + i10;
            const float* q11 = in + (size_t)cin0 * HW + i11;

            // B fragment: lane = column N (its pixel), 16 cin samples of this tap
            v16h bfrag;
#pragma unroll
            for (int c = 0; c < 16; ++c) {
                size_t cs = (size_t)c * HW;
                float v = w00 * q00[cs] + w01 * q01[cs] + w10 * q10[cs] + w11 * q11[cs];
                bfrag[c] = (_Float16)v;
            }

            // A fragments: lane holds row M = mt*16 + (lane&15);
            // elements 0..15 <- K off khalf*16 + (0..15)
            const _Float16* wbase = Wh + kb * 32 + khalf * 16;
#pragma unroll
            for (int mt = 0; mt < 4; ++mt) {
                const _Float16* wp = wbase + (size_t)(mt * 16 + (lane & 15)) * KDIM;
                v8h alo = *(const v8h*)(wp);
                v8h ahi = *(const v8h*)(wp + 8);
                v16h afrag;
#pragma unroll
                for (int i = 0; i < 8; ++i) { afrag[i] = alo[i]; afrag[8 + i] = ahi[i]; }

                acc[mt] = __builtin_amdgcn_wmma_f32_16x16x32_f16(
                    false, afrag, false, bfrag, (short)0, acc[mt], false, false);
            }
        }
    }

    // Epilogue: fused bias+BN (scale/shift) + ReLU, D layout: M = r + 8*khalf
#pragma unroll
    for (int mt = 0; mt < 4; ++mt) {
#pragma unroll
        for (int r = 0; r < 8; ++r) {
            int o = mt * 16 + khalf * 8 + r;
            float val = fmaf(acc[mt][r], sc[o], sh[o]);
            out[(size_t)o * HW + pix] = fmaxf(val, 0.0f);
        }
    }
}

// ---------------------------------------------------------------------------
// Final 1x1 conv: 64 -> 1 channel. Memory bound; one thread per pixel.
// ---------------------------------------------------------------------------
__global__ __launch_bounds__(256) void conv1x1_kernel(
    const float* __restrict__ in, const float* __restrict__ cw,
    const float* __restrict__ cb, float* __restrict__ out)
{
    int pix = blockIdx.x * blockDim.x + threadIdx.x;
    float a = cb[0];
#pragma unroll
    for (int c = 0; c < CIN; ++c)
        a = fmaf(cw[c], in[(size_t)c * HW + pix], a);
    out[pix] = a;
}

// ---------------------------------------------------------------------------
extern "C" void kernel_launch(void* const* d_in, const int* in_sizes, int n_in,
                              void* d_out, int out_size, void* d_ws, size_t ws_size,
                              hipStream_t stream)
{
    (void)in_sizes; (void)n_in; (void)out_size; (void)ws_size;

    const float* x      = (const float*)d_in[0];
    const float* o1w    = (const float*)d_in[1];
    const float* o1b    = (const float*)d_in[2];
    const float* d1w    = (const float*)d_in[3];
    const float* d1b    = (const float*)d_in[4];
    const float* g1     = (const float*)d_in[5];
    const float* b1     = (const float*)d_in[6];
    const float* m1     = (const float*)d_in[7];
    const float* v1     = (const float*)d_in[8];
    const float* o2w    = (const float*)d_in[9];
    const float* o2b    = (const float*)d_in[10];
    const float* d2w    = (const float*)d_in[11];
    const float* d2b    = (const float*)d_in[12];
    const float* g2     = (const float*)d_in[13];
    const float* b2     = (const float*)d_in[14];
    const float* m2     = (const float*)d_in[15];
    const float* v2     = (const float*)d_in[16];
    const float* cw     = (const float*)d_in[17];
    const float* cb     = (const float*)d_in[18];
    float* out = (float*)d_out;

    // workspace carve-up (all chunks are multiples of 256 B)
    char* p = (char*)d_ws;
    _Float16* Wh1 = (_Float16*)p; p += (size_t)64 * KDIM * sizeof(_Float16);
    _Float16* Wh2 = (_Float16*)p; p += (size_t)64 * KDIM * sizeof(_Float16);
    float* sc1 = (float*)p; p += 64 * sizeof(float);
    float* sh1 = (float*)p; p += 64 * sizeof(float);
    float* sc2 = (float*)p; p += 64 * sizeof(float);
    float* sh2 = (float*)p; p += 64 * sizeof(float);
    float* off1 = (float*)p; p += (size_t)18 * HW * sizeof(float);
    float* h1   = (float*)p; p += (size_t)64 * HW * sizeof(float);
    float* off2 = (float*)p; p += (size_t)18 * HW * sizeof(float);
    float* h2   = (float*)p; p += (size_t)64 * HW * sizeof(float);

    const int PIX_BLOCKS = HW / 256;                 // 1024
    const int DEF_BLOCKS = (HW / 16) * 32 / 256;     // 2048 (one wave per 16 px)

    prep_kernel<<<144, 256, 0, stream>>>(d1w, d1b, g1, b1, m1, v1,
                                         d2w, d2b, g2, b2, m2, v2,
                                         Wh1, Wh2, sc1, sh1, sc2, sh2);

    offconv_kernel<<<PIX_BLOCKS, 256, 0, stream>>>(x, o1w, o1b, off1);
    deform_wmma_kernel<<<DEF_BLOCKS, 256, 0, stream>>>(x, off1, Wh1, sc1, sh1, h1);
    offconv_kernel<<<PIX_BLOCKS, 256, 0, stream>>>(h1, o2w, o2b, off2);
    deform_wmma_kernel<<<DEF_BLOCKS, 256, 0, stream>>>(h1, off2, Wh2, sc2, sh2, h2);
    conv1x1_kernel<<<PIX_BLOCKS, 256, 0, stream>>>(h2, cw, cb, out);
}
